// TriangleMultiplicativeUpdate_19490561589259
// MI455X (gfx1250) — matrix operand order repaired and checked by reading, not compiled, not run on hardware
//
#include <hip/hip_runtime.h>

typedef _Float16 f16;
typedef __attribute__((ext_vector_type(4)))  _Float16 v4h;
typedef __attribute__((ext_vector_type(8)))  _Float16 v8h;
typedef __attribute__((ext_vector_type(16))) _Float16 v16h;
typedef __attribute__((ext_vector_type(8)))  float    v8f;

#define NN 384
#define CC 128
#define NTILES (NN*NN/16)   // 9216
#define TPI (NN/16)         // 24 tiles per i-row
#define WPB 8               // waves per block (256 threads)

// Weight/fragment workspace layout (d_ws):
//   [0, 6*32768)        : f16 B-fragments, matrix order {ap, ag, bp, bg, g, z}
//                         per matrix: 32 fragments (kc*8+nc) of 1024B;
//                         fragment = plane0[lane*16B] + plane1(+512B)[lane*16B]
//   [196608, 196608+3072): folded f32 biases, 6 * 128
#define FRAG_BYTES (6*32768)

__device__ __forceinline__ v8f wmma32(v16h a, v16h b, v8f c) {
  return __builtin_amdgcn_wmma_f32_16x16x32_f16(false, a, false, b, (short)0, c,
                                                false, false);
}
__device__ __forceinline__ float sigm(float x) { return 1.0f / (1.0f + __expf(-x)); }

// ---------------------------------------------------------------------------
// Prep: fold LN affine into weights/biases, pre-swizzle weights into the
// WMMA B-operand fragment layout (f16), write to global workspace.
// B layout (16x16x32 f16): lane l -> n = l%16, h = l/16; slot e<8 -> k = 32*kc + 8*h + e, e>=8 -> k = 32*kc + 16 + 8*h + (e-8)
// ---------------------------------------------------------------------------
__global__ void prep_kernel(const float* w_ap, const float* w_ag, const float* w_bp,
                            const float* w_bg, const float* w_g,  const float* w_z,
                            const float* b_ap, const float* b_ag, const float* b_bp,
                            const float* b_bg, const float* b_g,  const float* b_z,
                            const float* ln_in_w, const float* ln_in_b,
                            const float* ln_out_w, const float* ln_out_b,
                            f16* frag_out, float* bias_out) {
  const float* Ws[6] = {w_ap, w_ag, w_bp, w_bg, w_g, w_z};
  const float* Bs[6] = {b_ap, b_ag, b_bp, b_bg, b_g, b_z};
  int tid = threadIdx.x;

  // Fragment swizzle, 4 elements (one float4 of k) at a time.
  for (int idx = tid; idx < 6 * 128 * 32; idx += blockDim.x) {
    int p   = idx / (128 * 32);
    int rem = idx % (128 * 32);
    int n   = rem / 32;
    int k4  = (rem % 32) * 4;
    const float* lnw = (p < 5) ? ln_in_w : ln_out_w;
    float4 w = *(const float4*)(Ws[p] + n * 128 + k4);
    float4 l = *(const float4*)(lnw + k4);
    v4h hv;
    hv[0] = (f16)(w.x * l.x); hv[1] = (f16)(w.y * l.y);
    hv[2] = (f16)(w.z * l.z); hv[3] = (f16)(w.w * l.w);
    int kc  = k4 >> 5;
    int hh  = (k4 >> 3) & 1;
    int s   = (k4 & 7) + 8 * ((k4 >> 4) & 1);  // slot base; s%8 in {0,4}
    int lam = (n & 15) + 16 * hh;
    int nc  = n >> 4;
    size_t off = (size_t)p * 32768 + (size_t)(kc * 8 + nc) * 1024 +
                 ((s >= 8) ? 512 : 0) + lam * 16 + (s & 7) * 2;
    *(v4h*)((char*)frag_out + off) = hv;
  }

  // Folded biases: b'[n] = b[n] + sum_k W[n,k]*ln_b[k]
  for (int idx = tid; idx < 6 * 128; idx += blockDim.x) {
    int p = idx / 128, n = idx % 128;
    const float* lnb = (p < 5) ? ln_in_b : ln_out_b;
    float acc = Bs[p][n];
    for (int k = 0; k < 128; k++) acc += Ws[p][n * 128 + k] * lnb[k];
    bias_out[idx] = acc;
  }
}

// ---------------------------------------------------------------------------
// Fused triangle-multiplicative-update kernel.
// ---------------------------------------------------------------------------
__launch_bounds__(256)
__global__ void tmu_kernel(const float* __restrict__ z, const float* __restrict__ mask,
                           const f16* __restrict__ frags, const float* __restrict__ biasv,
                           float* __restrict__ out) {
  __shared__ __align__(16) f16 sW[4 * 16384];      // ap, ag, bp, bg fragments: 128 KB
  __shared__ __align__(16) f16 sX[WPB][2048];      // per-wave x fragment scratch: 32 KB

  int tid = threadIdx.x;

  // Stage the 4 projection weight matrices (pre-swizzled) into LDS.
  {
    const uint4* src = (const uint4*)frags;
    uint4* dst = (uint4*)sW;
    for (int i = tid; i < 8192; i += blockDim.x) dst[i] = src[i];
  }
  __syncthreads();

  const int lane = tid & 31;
  const int wv   = tid >> 5;
  const int l    = lane & 15;   // N / row-in-half index
  const int h    = lane >> 4;   // lane half

  const f16* wgf = frags + 4 * 16384;  // w_g fragments (global, L1/L2-hot)
  const f16* wzf = frags + 5 * 16384;  // w_z fragments

  // LDS B-fragment fetch (2x contiguous b128 per lane)
  auto ldsB = [&](int p, int kc, int nc) -> v16h {
    const f16* b = sW + p * 16384 + (kc * 8 + nc) * 512 + lane * 8;
    v8h lo = *(const v8h*)(b);
    v8h hi = *(const v8h*)(b + 256);
    return __builtin_shufflevector(lo, hi, 0,1,2,3,4,5,6,7,8,9,10,11,12,13,14,15);
  };
  // Global B-fragment fetch for w_g / w_z
  auto glbB = [&](const f16* mat, int kc, int nc) -> v16h {
    const f16* b = mat + (kc * 8 + nc) * 512 + lane * 8;
    v8h lo = *(const v8h*)(b);
    v8h hi = *(const v8h*)(b + 256);
    return __builtin_shufflevector(lo, hi, 0,1,2,3,4,5,6,7,8,9,10,11,12,13,14,15);
  };

  // Load 16 z-rows, LayerNorm (affine folded into weights), emit f16 A-frags.
  // A layout: slots 0-7 -> k = 32*kc + 8*h + e ; slots 8-15 -> k = 32*kc + 16 + 8*h + (e-8)
  auto makeA = [&](int rr, v16h* A) {
    float vals[4][2][8];
    float s = 0.f, ss = 0.f;
    const float* zp = z + (size_t)rr * CC;
#pragma unroll
    for (int kc = 0; kc < 4; kc++) {
#pragma unroll
      for (int u = 0; u < 2; u++) {
        int c0 = 32 * kc + 16 * u + 8 * h;
        float4 a4 = *(const float4*)(zp + c0);
        float4 b4 = *(const float4*)(zp + c0 + 4);
        vals[kc][u][0] = a4.x; vals[kc][u][1] = a4.y;
        vals[kc][u][2] = a4.z; vals[kc][u][3] = a4.w;
        vals[kc][u][4] = b4.x; vals[kc][u][5] = b4.y;
        vals[kc][u][6] = b4.z; vals[kc][u][7] = b4.w;
        s  += a4.x + a4.y + a4.z + a4.w + b4.x + b4.y + b4.z + b4.w;
        ss += a4.x*a4.x + a4.y*a4.y + a4.z*a4.z + a4.w*a4.w +
              b4.x*b4.x + b4.y*b4.y + b4.z*b4.z + b4.w*b4.w;
      }
    }
    s  += __shfl_xor(s, 16, 32);   // lane^16 holds the complementary 64 cols of same row
    ss += __shfl_xor(ss, 16, 32);
    float mean = s * (1.f / 128.f);
    float var  = ss * (1.f / 128.f) - mean * mean;
    float inv  = rsqrtf(var + 1e-5f);
#pragma unroll
    for (int kc = 0; kc < 4; kc++) {
      v16h av;
#pragma unroll
      for (int e = 0; e < 8; e++) {
        av[e]     = (f16)((vals[kc][0][e] - mean) * inv);
        av[e + 8] = (f16)((vals[kc][1][e] - mean) * inv);
      }
      A[kc] = av;
    }
  };

  const int numWaves = gridDim.x * WPB;
  const int wgid = blockIdx.x * WPB + wv;

  for (int t = wgid; t < NTILES; t += numWaves) {
    const int i0 = t / TPI;
    const int j0 = (t % TPI) * 16;
    const int r0 = t * 16;                 // == i0*NN + j0
    const int r  = r0 + l;                 // this lane's LN row (i0, j0+l)
    const int rT = (j0 + l) * NN + i0;     // transposed row (j0+l, i0)

    v16h AR[4], AT[4];
    makeA(r,  AR);   // zn on (i,j) rows  -> ap, ag, g
    makeA(rT, AT);   // zn on (j,i) rows  -> bp, bg

    // per-output-row combined mask: mask[i0,j0+row] * mask[j0+row,i0]
    float mm8[8];
#pragma unroll
    for (int v = 0; v < 8; v++) {
      int row = 8 * h + v;
      mm8[v] = mask[(size_t)r0 + row] * mask[(size_t)(j0 + row) * NN + i0];
    }

    // ---- Phase 1: four projections, gate, x = a*b*mask -> LDS (A-frag layout)
    // B-fragment DS loads are double-buffered across kc so WMMAs overlap LDS
    // latency (partial s_wait_dscnt instead of full drains).
    f16* xs = sX[wv];
#pragma unroll
    for (int nc = 0; nc < 8; nc++) {
      v8f accA = {}, accAg = {}, accB = {}, accBg = {};
      v16h c0 = ldsB(0, 0, nc), c1 = ldsB(1, 0, nc);
      v16h c2 = ldsB(2, 0, nc), c3 = ldsB(3, 0, nc);
      // kc = 0 (prefetch kc=1)
      v16h n0 = ldsB(0, 1, nc), n1 = ldsB(1, 1, nc);
      v16h n2 = ldsB(2, 1, nc), n3 = ldsB(3, 1, nc);
      accA  = wmma32(AR[0], c0, accA);
      accAg = wmma32(AR[0], c1, accAg);
      accB  = wmma32(AT[0], c2, accB);
      accBg = wmma32(AT[0], c3, accBg);
      // kc = 1 (prefetch kc=2)
      c0 = ldsB(0, 2, nc); c1 = ldsB(1, 2, nc);
      c2 = ldsB(2, 2, nc); c3 = ldsB(3, 2, nc);
      accA  = wmma32(AR[1], n0, accA);
      accAg = wmma32(AR[1], n1, accAg);
      accB  = wmma32(AT[1], n2, accB);
      accBg = wmma32(AT[1], n3, accBg);
      // kc = 2 (prefetch kc=3)
      n0 = ldsB(0, 3, nc); n1 = ldsB(1, 3, nc);
      n2 = ldsB(2, 3, nc); n3 = ldsB(3, 3, nc);
      accA  = wmma32(AR[2], c0, accA);
      accAg = wmma32(AR[2], c1, accAg);
      accB  = wmma32(AT[2], c2, accB);
      accBg = wmma32(AT[2], c3, accBg);
      // kc = 3
      accA  = wmma32(AR[3], n0, accA);
      accAg = wmma32(AR[3], n1, accAg);
      accB  = wmma32(AT[3], n2, accB);
      accBg = wmma32(AT[3], n3, accBg);

      int n = nc * 16 + l;
      float bA  = biasv[0 * 128 + n], bAg = biasv[1 * 128 + n];
      float bB  = biasv[2 * 128 + n], bBg = biasv[3 * 128 + n];
      // C/D layout: lane holds rows M = 8*h + v, col n
      int u   = (n >> 4) & 1;       // (col % 32) >= 16 ?
      int hp  = (l >> 3);           // dest lane half
      int sl  = u * 8 + (l & 7);    // dest slot
      int kcD = n >> 5;
#pragma unroll
      for (int v = 0; v < 8; v++) {
        float a = (accA[v] + bA) * sigm(accAg[v] + bAg);
        float b = (accB[v] + bB) * sigm(accBg[v] + bBg);
        float x = a * b * mm8[v];
        int lamp = (8 * h + v) + 16 * hp;
        xs[kcD * 512 + ((sl >= 8) ? 256 : 0) + lamp * 8 + (sl & 7)] = (f16)x;
      }
    }

    // ---- Phase 2: LayerNorm(x) straight out of the fragment-layout scratch.
    // Lane's region *is* row (lane%16), half-h column pattern: read back as frags.
    v16h XF[4];
    float s2 = 0.f, ss2 = 0.f;
#pragma unroll
    for (int kc = 0; kc < 4; kc++) {
      const f16* bp = xs + kc * 512;
      v8h lo = *(const v8h*)(bp + lane * 8);
      v8h hi = *(const v8h*)(bp + 256 + lane * 8);
      XF[kc] = __builtin_shufflevector(lo, hi, 0,1,2,3,4,5,6,7,8,9,10,11,12,13,14,15);
#pragma unroll
      for (int e = 0; e < 16; e++) {
        float xv = (float)XF[kc][e];
        s2 += xv; ss2 += xv * xv;
      }
    }
    s2  += __shfl_xor(s2, 16, 32);
    ss2 += __shfl_xor(ss2, 16, 32);
    float mean2 = s2 * (1.f / 128.f);
    float var2  = ss2 * (1.f / 128.f) - mean2 * mean2;
    float inv2  = rsqrtf(var2 + 1e-5f);
#pragma unroll
    for (int kc = 0; kc < 4; kc++) {
#pragma unroll
      for (int e = 0; e < 16; e++)
        XF[kc][e] = (f16)(((float)XF[kc][e] - mean2) * inv2);
    }

    // ---- Phase 3: out = (LN(x) @ w_z'^T + b_z') * sigmoid(zn @ w_g'^T + b_g')
    // Global fragment loads double-buffered across kc (LOADcnt overlap).
#pragma unroll
    for (int nc = 0; nc < 8; nc++) {
      v8f accZ = {}, accG = {};
      v16h zc = glbB(wzf, 0, nc), gc = glbB(wgf, 0, nc);
      v16h zn = glbB(wzf, 1, nc), gn = glbB(wgf, 1, nc);
      accZ = wmma32(XF[0], zc, accZ);
      accG = wmma32(AR[0], gc, accG);
      zc = glbB(wzf, 2, nc); gc = glbB(wgf, 2, nc);
      accZ = wmma32(XF[1], zn, accZ);
      accG = wmma32(AR[1], gn, accG);
      zn = glbB(wzf, 3, nc); gn = glbB(wgf, 3, nc);
      accZ = wmma32(XF[2], zc, accZ);
      accG = wmma32(AR[2], gc, accG);
      accZ = wmma32(XF[3], zn, accZ);
      accG = wmma32(AR[3], gn, accG);

      int n = nc * 16 + l;
      float bz = biasv[5 * 128 + n];
      float bg = biasv[4 * 128 + n];
#pragma unroll
      for (int v = 0; v < 8; v++) {
        float o = (accZ[v] + bz) * sigm(accG[v] + bg);
        out[(size_t)(r0 + 8 * h + v) * CC + n] = o;
      }
    }
  }
}

// ---------------------------------------------------------------------------
extern "C" void kernel_launch(void* const* d_in, const int* in_sizes, int n_in,
                              void* d_out, int out_size, void* d_ws, size_t ws_size,
                              hipStream_t stream) {
  (void)in_sizes; (void)n_in; (void)out_size; (void)ws_size;
  const float* z        = (const float*)d_in[0];
  const float* mask     = (const float*)d_in[1];
  const float* w_ap     = (const float*)d_in[2];
  const float* b_ap     = (const float*)d_in[3];
  const float* w_bp     = (const float*)d_in[4];
  const float* b_bp     = (const float*)d_in[5];
  const float* w_ag     = (const float*)d_in[6];
  const float* b_ag     = (const float*)d_in[7];
  const float* w_bg     = (const float*)d_in[8];
  const float* b_bg     = (const float*)d_in[9];
  const float* w_g      = (const float*)d_in[10];
  const float* b_g      = (const float*)d_in[11];
  const float* w_z      = (const float*)d_in[12];
  const float* b_z      = (const float*)d_in[13];
  const float* ln_in_w  = (const float*)d_in[14];
  const float* ln_in_b  = (const float*)d_in[15];
  const float* ln_out_w = (const float*)d_in[16];
  const float* ln_out_b = (const float*)d_in[17];

  f16*   frag_ws = (f16*)d_ws;
  float* bias_ws = (float*)((char*)d_ws + FRAG_BYTES);

  prep_kernel<<<1, 256, 0, stream>>>(w_ap, w_ag, w_bp, w_bg, w_g, w_z,
                                     b_ap, b_ag, b_bp, b_bg, b_g, b_z,
                                     ln_in_w, ln_in_b, ln_out_w, ln_out_b,
                                     frag_ws, bias_ws);
  tmu_kernel<<<384, 256, 0, stream>>>(z, mask, frag_ws, bias_ws, (float*)d_out);
}
